// RegLSTM_57715770524202
// MI455X (gfx1250) — compile-verified, matrix operand order checked by reading, not run on hardware
//
#include <hip/hip_runtime.h>
#include <hip/hip_bf16.h>
#include <stdint.h>

// Problem dims
#define B_    64
#define T_    512
#define E_    300
#define EP_   320      // embed K padded to multiple of 32
#define H_    512
#define NCLS  10

// Persistent-kernel partitioning
#define NWG   16       // one workgroup per WGP slice of hidden units
#define HS    32       // hidden units per WG
#define GC    128      // gate columns per WG (4 * HS)

// LDS strides (in elements) -- padded for bank-conflict-free fragment rows
#define WHH_STRIDE 520
#define WIH_STRIDE 328
#define H_STRIDE   520
#define GATE_STRIDE 132

// LDS layout (bytes)
#define OFF_WHH  0
#define OFF_WIH  133120          // 128*520*2
#define OFF_H    217088          // + 128*328*2
#define OFF_G    283648          // + 64*520*2
#define OFF_B    317440          // + 64*132*4
#define SMEM_BYTES 317952        // + 128*4

typedef __attribute__((ext_vector_type(16))) __bf16 bf16x16;
typedef __attribute__((ext_vector_type(8)))  float  f32x8;

union FragU { bf16x16 v; struct { uint4 lo, hi; } q; };
union H8U   { uint4 q; __bf16 h[8]; };

// Load one 16x32 bf16 fragment half-row pair per lane (two b128 loads).
// Caller passes p = base + row*stride + kbase + ((lane>>4)<<3).
__device__ __forceinline__ bf16x16 load_frag(const __bf16* p) {
  FragU f;
  f.q.lo = *reinterpret_cast<const uint4*>(p);
  f.q.hi = *reinterpret_cast<const uint4*>(p + 16);
  return f.v;
}

__device__ __forceinline__ float sigm(float x) { return 1.f / (1.f + __expf(-x)); }
__device__ __forceinline__ float tanh_f(float x) {
  float e = __expf(-2.f * fabsf(x));
  float t = (1.f - e) / (1.f + e);
  return x >= 0.f ? t : -t;
}

// CDNA5 async global->LDS copy (16B per lane), tracked by ASYNCcnt.
// VDST carries the workgroup-relative LDS byte address (low 32 bits of the
// generic shared pointer per the flat-aperture mapping); hardware adds LDS_BASE.
__device__ __forceinline__ void async_g2l_b128(unsigned lds_off,
                                               unsigned long long gaddr) {
  asm volatile("global_load_async_to_lds_b128 %0, %1, off"
               :: "v"(lds_off), "v"(gaddr)
               : "memory");
}
__device__ __forceinline__ void wait_asynccnt0() {
  asm volatile("s_wait_asynccnt 0x0" ::: "memory");
}

// ---------------------------------------------------------------------------
// Kernel 1: embedding gather + f32->bf16 + zero-pad K to 320
// ---------------------------------------------------------------------------
__global__ void embed_gather(const int* __restrict__ tokens,
                             const float* __restrict__ embed_w,
                             __bf16* __restrict__ x_emb) {
  size_t gid = (size_t)blockIdx.x * blockDim.x + threadIdx.x;
  const size_t total = (size_t)B_ * T_ * EP_;
  if (gid >= total) return;
  int e = (int)(gid % EP_);
  size_t row = gid / EP_;
  int tok = tokens[row];
  float v = (e < E_) ? embed_w[(size_t)tok * E_ + e] : 0.f;
  x_emb[gid] = (__bf16)v;
}

// ---------------------------------------------------------------------------
// Kernel 2: persistent LSTM. 16 WGs x 256 threads (8 waves), weights in LDS.
// ---------------------------------------------------------------------------
__global__ void __launch_bounds__(256, 1)
lstm_persistent(const float* __restrict__ w_ih, const float* __restrict__ w_hh,
                const float* __restrict__ b_ih, const float* __restrict__ b_hh,
                const __bf16* __restrict__ x_emb,
                __bf16* __restrict__ h_g,          // [2][B_*H_] ping-pong
                float* __restrict__ pooled,        // [B_*H_]
                unsigned int* __restrict__ bar_cnt) {
  extern __shared__ char smem[];
  __bf16* sWhh = (__bf16*)(smem + OFF_WHH);   // [GC][WHH_STRIDE] bf16
  __bf16* sWih = (__bf16*)(smem + OFF_WIH);   // [GC][WIH_STRIDE] bf16
  __bf16* sH   = (__bf16*)(smem + OFF_H);     // [B_][H_STRIDE]   bf16
  float*  sG   = (float*) (smem + OFF_G);     // [B_][GATE_STRIDE] f32
  float*  sB   = (float*) (smem + OFF_B);     // [GC] combined bias

  const int tid  = threadIdx.x;
  const int hid0 = blockIdx.x * HS;

  // ---- stage weight slices into LDS as bf16 ----
  for (int i = tid; i < GC * H_; i += 256) {
    int r = i >> 9, k = i & (H_ - 1);
    int grow = ((r >> 5) << 9) + hid0 + (r & 31);    // i/f/g/o gate row
    sWhh[r * WHH_STRIDE + k] = (__bf16)w_hh[(size_t)grow * H_ + k];
  }
  for (int i = tid; i < GC * EP_; i += 256) {
    int r = i / EP_, k = i - r * EP_;
    int grow = ((r >> 5) << 9) + hid0 + (r & 31);
    sWih[r * WIH_STRIDE + k] = (k < E_) ? (__bf16)w_ih[(size_t)grow * E_ + k] : (__bf16)0.f;
  }
  for (int i = tid; i < GC; i += 256) {
    int grow = ((i >> 5) << 9) + hid0 + (i & 31);
    sB[i] = b_ih[grow] + b_hh[grow];
  }
  for (int i = tid; i < B_ * H_STRIDE; i += 256) sH[i] = (__bf16)0.f;  // h(-1)=0
  __syncthreads();

  // ---- wave -> tile mapping: wave w owns m-tile (w>>1), n-tiles 4*(w&1).. ----
  const int lane = tid & 31;
  const int wv   = tid >> 5;
  const int mt   = wv >> 1;
  const int nb   = (wv & 1) * 4;
  const int rA   = mt * 16 + (lane & 15);     // batch row for A fragments
  const int hsel = (lane >> 4) << 3;          // K sub-offset 0 / 8 per ISA layout

  int   nrow[4];
  float biasv[4];
  #pragma unroll
  for (int ni = 0; ni < 4; ++ni) {
    nrow[ni]  = (nb + ni) * 16 + (lane & 15);
    biasv[ni] = sB[nrow[ni]];
  }
  const __bf16* bHbase[4];
  const __bf16* bXbase[4];
  #pragma unroll
  for (int ni = 0; ni < 4; ++ni) {
    bHbase[ni] = sWhh + nrow[ni] * WHH_STRIDE + hsel;
    bXbase[ni] = sWih + nrow[ni] * WIH_STRIDE + hsel;
  }

  // combine-phase mapping: thread -> (batch, 8 hidden units)
  const int cb = tid >> 2;
  const int cu = (tid & 3) << 3;
  float c_reg[8], p_reg[8];
  #pragma unroll
  for (int j = 0; j < 8; ++j) { c_reg[j] = 0.f; p_reg[j] = 0.f; }

  unsigned int bar_target = 0;

  for (int t = 0; t < T_; ++t) {
    f32x8 acc[4];
    #pragma unroll
    for (int ni = 0; ni < 4; ++ni)
      #pragma unroll
      for (int r = 0; r < 8; ++r) acc[ni][r] = 0.f;

    const __bf16* aH = sH + rA * H_STRIDE + hsel;
    const __bf16* aX = x_emb + ((size_t)rA * T_ + t) * EP_ + hsel;
    if (t + 1 < T_) __builtin_prefetch(aX + EP_, 0, 0);  // next x row -> L2/L0

    // ---- recurrent part: h[64x512] @ w_hh_sliceT, K=512, double-buffered ----
    bf16x16 aF[2];
    bf16x16 bF[2][4];
    aF[0] = load_frag(aH);
    #pragma unroll
    for (int ni = 0; ni < 4; ++ni) bF[0][ni] = load_frag(bHbase[ni]);

    #pragma unroll
    for (int kc = 0; kc < 16; ++kc) {
      const int cur = kc & 1, nxt = cur ^ 1;
      if (kc < 15) {                                // prefetch chunk kc+1
        aF[nxt] = load_frag(aH + (kc + 1) * 32);
        #pragma unroll
        for (int ni = 0; ni < 4; ++ni)
          bF[nxt][ni] = load_frag(bHbase[ni] + (kc + 1) * 32);
      }
      #pragma unroll
      for (int ni = 0; ni < 4; ++ni)
        acc[ni] = __builtin_amdgcn_wmma_f32_16x16x32_bf16(
            false, aF[cur], false, bF[cur][ni], (short)0, acc[ni], false, false);
    }

    // ---- input part: x_t[64x320] @ w_ih_sliceT, K=320, double-buffered ----
    aF[0] = load_frag(aX);                           // global (LOADcnt)
    #pragma unroll
    for (int ni = 0; ni < 4; ++ni) bF[0][ni] = load_frag(bXbase[ni]);

    #pragma unroll
    for (int kc = 0; kc < 10; ++kc) {
      const int cur = kc & 1, nxt = cur ^ 1;
      if (kc < 9) {
        aF[nxt] = load_frag(aX + (kc + 1) * 32);
        #pragma unroll
        for (int ni = 0; ni < 4; ++ni)
          bF[nxt][ni] = load_frag(bXbase[ni] + (kc + 1) * 32);
      }
      #pragma unroll
      for (int ni = 0; ni < 4; ++ni)
        acc[ni] = __builtin_amdgcn_wmma_f32_16x16x32_bf16(
            false, aF[cur], false, bF[cur][ni], (short)0, acc[ni], false, false);
    }

    // ---- bias + activation -> sG (C/D layout: M = r + 8*hi, N = lane&15) ----
    #pragma unroll
    for (int ni = 0; ni < 4; ++ni) {
      int n   = nb + ni;
      int col = n * 16 + (lane & 15);
      int gt  = n >> 1;                       // 0:i 1:f 2:g 3:o
      #pragma unroll
      for (int r = 0; r < 8; ++r) {
        int bb  = mt * 16 + r + ((lane >> 4) << 3);
        float x = acc[ni][r] + biasv[ni];
        sG[bb * GATE_STRIDE + col] = (gt == 2) ? tanh_f(x) : sigm(x);
      }
    }
    __syncthreads();

    // ---- pointwise cell update, relu-max pooling, write h slice ----
    const int hb = t & 1;
    H8U hh;
    #pragma unroll
    for (int j = 0; j < 8; ++j) {
      int u = cu + j;
      float iv = sG[cb * GATE_STRIDE +       u];
      float fv = sG[cb * GATE_STRIDE +  32 + u];
      float gv = sG[cb * GATE_STRIDE +  64 + u];
      float ov = sG[cb * GATE_STRIDE +  96 + u];
      float c  = fv * c_reg[j] + iv * gv;
      c_reg[j] = c;
      float h  = ov * tanh_f(c);
      p_reg[j] = fmaxf(p_reg[j], h);          // p>=0, so relu folds in
      hh.h[j]  = (__bf16)h;
    }
    *reinterpret_cast<uint4*>(h_g + (size_t)hb * B_ * H_ + cb * H_ + hid0 + cu) = hh.q;

    // ---- device-wide barrier (16 persistent WGs) ----
    bar_target += NWG;
    __syncthreads();
    if (tid == 0) {
      __threadfence();
      atomicAdd(bar_cnt, 1u);
      volatile unsigned int* vc = bar_cnt;
      while (*vc < bar_target) __builtin_amdgcn_s_sleep(1);
    }
    __syncthreads();
    __threadfence();

    // ---- broadcast full h(t) into LDS via CDNA5 async global->LDS DMA ----
    const __bf16* srcg = h_g + (size_t)hb * B_ * H_;
    #pragma unroll 4
    for (int i = tid; i < (B_ * H_) / 8; i += 256) {
      int row = i >> 6, qq = i & 63;
      unsigned lds_off =
          (unsigned)(uintptr_t)(sH + row * H_STRIDE + qq * 8);
      unsigned long long gaddr =
          (unsigned long long)(uintptr_t)(srcg + (size_t)i * 8);
      async_g2l_b128(lds_off, gaddr);
    }
    wait_asynccnt0();
    __syncthreads();
  }

  // ---- write pooled relu-max slice ----
  #pragma unroll
  for (int j = 0; j < 8; ++j)
    pooled[(size_t)cb * H_ + hid0 + cu + j] = p_reg[j];
}

// ---------------------------------------------------------------------------
// Kernel 3: FC (10x512) + softmax. d_out = [probs(640) | scores(640)] f32.
// ---------------------------------------------------------------------------
__global__ void fc_softmax(const float* __restrict__ pooled,
                           const float* __restrict__ fc1_w,
                           const float* __restrict__ fc1_b,
                           float* __restrict__ out) {
  __shared__ float sc[NCLS];
  int b = blockIdx.x;
  int c = threadIdx.x;
  if (c < NCLS) {
    float s = fc1_b[c];
    const float* pr = pooled + (size_t)b * H_;
    const float* wr = fc1_w + (size_t)c * H_;
    for (int h = 0; h < H_; ++h) s += pr[h] * wr[h];
    sc[c] = s;
  }
  __syncthreads();
  if (c < NCLS) {
    float m = sc[0];
    for (int i = 1; i < NCLS; ++i) m = fmaxf(m, sc[i]);
    float sum = 0.f;
    for (int i = 0; i < NCLS; ++i) sum += __expf(sc[i] - m);
    out[(size_t)b * NCLS + c] = __expf(sc[c] - m) / sum;   // probs
    out[(size_t)B_ * NCLS + (size_t)b * NCLS + c] = sc[c]; // scores
  }
}

// ---------------------------------------------------------------------------
extern "C" void kernel_launch(void* const* d_in, const int* in_sizes, int n_in,
                              void* d_out, int out_size, void* d_ws, size_t ws_size,
                              hipStream_t stream) {
  const int*   tokens  = (const int*)d_in[0];
  const float* embed_w = (const float*)d_in[1];
  const float* w_ih    = (const float*)d_in[2];
  const float* w_hh    = (const float*)d_in[3];
  const float* b_ih    = (const float*)d_in[4];
  const float* b_hh    = (const float*)d_in[5];
  const float* fc1_w   = (const float*)d_in[6];
  const float* fc1_b   = (const float*)d_in[7];
  float* out = (float*)d_out;

  // workspace layout (~21.3 MB)
  char* ws = (char*)d_ws;
  unsigned int* cnt = (unsigned int*)ws;                        // 128 B
  __bf16* x_emb = (__bf16*)(ws + 128);                          // 20,971,520 B
  __bf16* h_g   = (__bf16*)(ws + 128 + 20971520);               // 131,072 B
  float*  pooled = (float*)(ws + 128 + 20971520 + 131072);      // 131,072 B

  hipMemsetAsync(cnt, 0, 128, stream);   // deterministic barrier counter

  const size_t total = (size_t)B_ * T_ * EP_;
  embed_gather<<<(unsigned)((total + 255) / 256), 256, 0, stream>>>(tokens, embed_w, x_emb);
  lstm_persistent<<<NWG, 256, SMEM_BYTES, stream>>>(w_ih, w_hh, b_ih, b_hh,
                                                    x_emb, h_g, pooled, cnt);
  fc_softmax<<<B_, 32, 0, stream>>>(pooled, fc1_w, fc1_b, out);
}
